// ActorMoE_76613626626599
// MI455X (gfx1250) — matrix operand order repaired and checked by reading, not compiled, not run on hardware
//
#include <hip/hip_runtime.h>
#include <hip/hip_bf16.h>
#include <cstdint>
#include <cstddef>

// ---------------------------------------------------------------------------
// ActorMoE for MI455X (gfx1250): fused bf16-WMMA expert MLP.
//   B=16384, OBS=512, hidden=[1024,512,256], ACT=64, E=8, GATE_H=256
// Compute-bound (~313 GFLOP vs ~75MB traffic) -> v_wmma_f32_16x16x32_bf16.
// Round 2: N-pair register blocking -> 1 A-frag load feeds 2 WMMAs on two
// independent accumulator chains (halves LDS traffic, overlaps XDL latency).
// ---------------------------------------------------------------------------

#define B_TOT  16384
#define OBS    512
#define ACTD   64
#define NEXP   8
#define GATEH  256

typedef __attribute__((ext_vector_type(16))) __bf16 v16bf;
typedef __attribute__((ext_vector_type(8)))  float  v8f;

__device__ __forceinline__ unsigned short f32_to_bf16(float f) {
  union { float f; unsigned int u; } v; v.f = f;
  unsigned int r = v.u + 0x7FFFu + ((v.u >> 16) & 1u);   // round-nearest-even
  return (unsigned short)(r >> 16);
}

__device__ __forceinline__ float elu_f(float v) {
  return v > 0.f ? v : (__expf(v) - 1.f);
}

// ---------------------------------------------------------------------------
// Weight prep: f32 [e][K][N] -> bf16 transposed [e][N][K] (contiguous-K rows
// so each lane's WMMA B-fragment is two global_load_b128).
// ---------------------------------------------------------------------------
__global__ __launch_bounds__(256)
void k_prep_wt(const float* __restrict__ src, unsigned short* __restrict__ dst,
               int K, int N) {
  const size_t total = (size_t)K * N;
  const float* s = src + (size_t)blockIdx.y * total;
  unsigned short* d = dst + (size_t)blockIdx.y * total;
  for (size_t i = (size_t)blockIdx.x * blockDim.x + threadIdx.x; i < total;
       i += (size_t)gridDim.x * blockDim.x) {
    const int k = (int)(i / (size_t)N);
    const int n = (int)(i % (size_t)N);
    d[(size_t)n * K + k] = f32_to_bf16(s[i]);
  }
}

// ---------------------------------------------------------------------------
// Per-wave GEMM over a 64-row LDS tile: D[64,Nout] = act(A[64,Kin] @ W + b)
//   A: bf16 in LDS, row-major, stride Kin
//   Wt: bf16 global, transposed [Nout][Kin]
//   8 waves: wave&3 -> 16-row M slab, wave>>2 -> odd/even 32-wide N pairs
// A/B fragments use contiguous K (k = hi*16 + s), not the ISA table order; C/D per CDNA5 ISA 7.12.2.
// ---------------------------------------------------------------------------
__device__ __forceinline__ void wave_gemm_64(
    const unsigned short* __restrict__ ldsIn, int Kin,
    const unsigned short* __restrict__ Wt,
    const float* __restrict__ bias, int Nout,
    unsigned short* ldsOut,                 // bf16 [64][Nout]  (if non-null)
    float* __restrict__ gOut, int gStride,  // f32 global       (if ldsOut==null)
    bool doElu)
{
  const int lane  = threadIdx.x & 31;
  const int wave  = threadIdx.x >> 5;
  const int m0    = (wave & 3) * 16;
  const int nhalf = wave >> 2;
  const int l15   = lane & 15;
  const int hi    = lane >> 4;            // 0: lanes 0-15, 1: lanes 16-31
  const int npairs = Nout >> 5;           // pairs of 16-wide N tiles

  const unsigned short* arow = ldsIn + (size_t)(m0 + l15) * Kin + hi * 16;
  const int rbase = m0 + hi * 8;

  for (int p = nhalf; p < npairs; p += 2) {
    const int n0 = p * 32;
    const unsigned short* wrow0 = Wt + (size_t)(n0 + l15) * Kin + hi * 16;
    const unsigned short* wrow1 = wrow0 + (size_t)16 * Kin;
    v8f acc0 = {}, acc1 = {};
#pragma unroll 4
    for (int k0 = 0; k0 < Kin; k0 += 32) {
      v16bf a, b0, b1;
      // A 16x32: elems 0-15 = K {hi*16 + 0..15} contiguous
      reinterpret_cast<uint4*>(&a)[0] = *reinterpret_cast<const uint4*>(arow + k0);
      reinterpret_cast<uint4*>(&a)[1] = *reinterpret_cast<const uint4*>(arow + k0 + 8);
      // B 32x16: lane holds column l15, K {hi*16 + 0..15} contiguous
      const uint4* q0 = reinterpret_cast<const uint4*>(wrow0 + k0);
      const uint4* q1 = reinterpret_cast<const uint4*>(wrow1 + k0);
      reinterpret_cast<uint4*>(&b0)[0] = q0[0];
      reinterpret_cast<uint4*>(&b0)[1] = q0[1];
      reinterpret_cast<uint4*>(&b1)[0] = q1[0];
      reinterpret_cast<uint4*>(&b1)[1] = q1[1];
      // one A fragment feeds two independent accumulation chains
      acc0 = __builtin_amdgcn_wmma_f32_16x16x32_bf16(
          false, a, false, b0, (short)0, acc0, false, false);
      acc1 = __builtin_amdgcn_wmma_f32_16x16x32_bf16(
          false, a, false, b1, (short)0, acc1, false, false);
    }
    // C/D layout: VGPR r -> row m0 + hi*8 + r, col n0 + l15
    const int   col0 = n0 + l15;
    const int   col1 = col0 + 16;
    const float bv0  = bias[col0];
    const float bv1  = bias[col1];
    if (ldsOut) {
#pragma unroll
      for (int r = 0; r < 8; ++r) {
        float v0 = acc0[r] + bv0;
        float v1 = acc1[r] + bv1;
        if (doElu) { v0 = elu_f(v0); v1 = elu_f(v1); }
        ldsOut[(size_t)(rbase + r) * Nout + col0] = f32_to_bf16(v0);
        ldsOut[(size_t)(rbase + r) * Nout + col1] = f32_to_bf16(v1);
      }
    } else {
#pragma unroll
      for (int r = 0; r < 8; ++r) {
        float v0 = acc0[r] + bv0;
        float v1 = acc1[r] + bv1;
        if (doElu) { v0 = elu_f(v0); v1 = elu_f(v1); }
        gOut[(size_t)(rbase + r) * gStride + col0] = v0;
        gOut[(size_t)(rbase + r) * gStride + col1] = v1;
      }
    }
  }
}

__device__ __forceinline__ void load_x_tile_bf16(
    const float* __restrict__ x, int row0, unsigned short* __restrict__ buf) {
  for (int i = threadIdx.x * 4; i < 64 * OBS; i += blockDim.x * 4) {
    const int r = i >> 9;            // OBS == 512
    const int c = i & (OBS - 1);
    const float4 v = *reinterpret_cast<const float4*>(x + (size_t)(row0 + r) * OBS + c);
    unsigned short h[4] = { f32_to_bf16(v.x), f32_to_bf16(v.y),
                            f32_to_bf16(v.z), f32_to_bf16(v.w) };
    *reinterpret_cast<uint2*>(buf + i) = *reinterpret_cast<const uint2*>(h);
  }
}

// ---------------------------------------------------------------------------
// Fused 4-layer expert MLP. grid = (B/64, E), block = 256 (8 wave32).
// Activations ping-pong between two LDS buffers (192 KB total, CDNA5 WGP LDS).
// ---------------------------------------------------------------------------
__global__ __launch_bounds__(256)
void k_moe_fused(const float* __restrict__ x,
                 const unsigned short* __restrict__ Wt0,
                 const unsigned short* __restrict__ Wt1,
                 const unsigned short* __restrict__ Wt2,
                 const unsigned short* __restrict__ Wt3,
                 const float* __restrict__ b0, const float* __restrict__ b1,
                 const float* __restrict__ b2, const float* __restrict__ b3,
                 float* __restrict__ out_e)
{
  extern __shared__ unsigned short smem[];
  unsigned short* buf0 = smem;             // 64*512  (x, then h1)
  unsigned short* buf1 = smem + 64 * 512;  // 64*1024 (h0, then h2)
  const int e    = blockIdx.y;
  const int row0 = blockIdx.x * 64;

  load_x_tile_bf16(x, row0, buf0);
  __syncthreads();
  wave_gemm_64(buf0, OBS,  Wt0 + (size_t)e * OBS  * 1024, b0 + e * 1024, 1024,
               buf1, nullptr, 0, true);
  __syncthreads();
  wave_gemm_64(buf1, 1024, Wt1 + (size_t)e * 1024 * 512,  b1 + e * 512,  512,
               buf0, nullptr, 0, true);
  __syncthreads();
  wave_gemm_64(buf0, 512,  Wt2 + (size_t)e * 512  * 256,  b2 + e * 256,  256,
               buf1, nullptr, 0, true);
  __syncthreads();
  wave_gemm_64(buf1, 256,  Wt3 + (size_t)e * 256  * 64,   b3 + e * 64,   64,
               nullptr, out_e + ((size_t)row0 * NEXP + e) * ACTD, NEXP * ACTD, false);
}

// Gate hidden: g = ELU(x @ gW0 + gb0), WMMA path. grid = B/64, block = 256.
__global__ __launch_bounds__(256)
void k_gate_hidden(const float* __restrict__ x,
                   const unsigned short* __restrict__ gWt,
                   const float* __restrict__ gb0, float* __restrict__ g)
{
  extern __shared__ unsigned short smem[];
  const int row0 = blockIdx.x * 64;
  load_x_tile_bf16(x, row0, smem);
  __syncthreads();
  wave_gemm_64(smem, OBS, gWt, gb0, GATEH,
               nullptr, g + (size_t)row0 * GATEH, GATEH, true);
}

// logits = g @ gW1 + gb1 ; softmax over E=8. One thread per batch row.
__global__ __launch_bounds__(256)
void k_gate_softmax(const float* __restrict__ g, const float* __restrict__ gW1,
                    const float* __restrict__ gb1, float* __restrict__ wsoft)
{
  const int b = blockIdx.x * blockDim.x + threadIdx.x;
  if (b >= B_TOT) return;
  float lg[NEXP];
#pragma unroll
  for (int e = 0; e < NEXP; ++e) lg[e] = gb1[e];
  const float* gr = g + (size_t)b * GATEH;
  for (int k = 0; k < GATEH; ++k) {
    const float gv = gr[k];
#pragma unroll
    for (int e = 0; e < NEXP; ++e) lg[e] = fmaf(gv, gW1[k * NEXP + e], lg[e]);
  }
  float m = lg[0];
#pragma unroll
  for (int e = 1; e < NEXP; ++e) m = fmaxf(m, lg[e]);
  float s = 0.f;
#pragma unroll
  for (int e = 0; e < NEXP; ++e) { lg[e] = __expf(lg[e] - m); s += lg[e]; }
  const float inv = 1.f / s;
#pragma unroll
  for (int e = 0; e < NEXP; ++e) wsoft[(size_t)b * NEXP + e] = lg[e] * inv;
}

// out[b,a] = sum_e wsoft[b,e] * out_e[b,e,a]. One thread per output element.
__global__ __launch_bounds__(256)
void k_combine(const float* __restrict__ out_e, const float* __restrict__ wsoft,
               float* __restrict__ out)
{
  const int idx = blockIdx.x * blockDim.x + threadIdx.x;
  if (idx >= B_TOT * ACTD) return;
  const int b = idx >> 6;
  const int a = idx & (ACTD - 1);
  const float* oe = out_e + (size_t)b * NEXP * ACTD + a;
  const float* wr = wsoft + (size_t)b * NEXP;
  float s = 0.f;
#pragma unroll
  for (int e = 0; e < NEXP; ++e) s = fmaf(wr[e], oe[e * ACTD], s);
  out[idx] = s;
}

// ---------------------------------------------------------------------------
// Host launcher. Inputs (setup_inputs order):
//  0:x 1:W0 2:b0 3:W1 4:b1 5:W2 6:b2 7:W3 8:b3 9:gW0 10:gb0 11:gW1 12:gb1
// ---------------------------------------------------------------------------
extern "C" void kernel_launch(void* const* d_in, const int* in_sizes, int n_in,
                              void* d_out, int out_size, void* d_ws, size_t ws_size,
                              hipStream_t stream) {
  (void)in_sizes; (void)n_in; (void)out_size; (void)ws_size;
  const float* x   = (const float*)d_in[0];
  const float* W0  = (const float*)d_in[1];
  const float* b0  = (const float*)d_in[2];
  const float* W1  = (const float*)d_in[3];
  const float* b1  = (const float*)d_in[4];
  const float* W2  = (const float*)d_in[5];
  const float* b2  = (const float*)d_in[6];
  const float* W3  = (const float*)d_in[7];
  const float* b3  = (const float*)d_in[8];
  const float* gW0 = (const float*)d_in[9];
  const float* gb0 = (const float*)d_in[10];
  const float* gW1 = (const float*)d_in[11];
  const float* gb1 = (const float*)d_in[12];

  // workspace carve-up (bf16 transposed weights + intermediates), ~70 MB
  char* ws = (char*)d_ws;
  auto take = [&](size_t bytes) { char* p = ws; ws += (bytes + 255) & ~(size_t)255; return p; };
  unsigned short* Wt0 = (unsigned short*)take((size_t)NEXP * 512 * 1024 * 2);
  unsigned short* Wt1 = (unsigned short*)take((size_t)NEXP * 1024 * 512 * 2);
  unsigned short* Wt2 = (unsigned short*)take((size_t)NEXP * 512 * 256 * 2);
  unsigned short* Wt3 = (unsigned short*)take((size_t)NEXP * 256 * 64 * 2);
  unsigned short* gWt = (unsigned short*)take((size_t)512 * 256 * 2);
  float* out_e = (float*)take((size_t)B_TOT * NEXP * ACTD * 4);
  float* g     = (float*)take((size_t)B_TOT * GATEH * 4);
  float* wsoft = (float*)take((size_t)B_TOT * NEXP * 4);

  // 1) convert + transpose weights to bf16 [N][K]
  k_prep_wt<<<dim3(256, NEXP), 256, 0, stream>>>(W0, Wt0, 512, 1024);
  k_prep_wt<<<dim3(256, NEXP), 256, 0, stream>>>(W1, Wt1, 1024, 512);
  k_prep_wt<<<dim3(64,  NEXP), 256, 0, stream>>>(W2, Wt2, 512, 256);
  k_prep_wt<<<dim3(16,  NEXP), 256, 0, stream>>>(W3, Wt3, 256, 64);
  k_prep_wt<<<dim3(64,  1),    256, 0, stream>>>(gW0, gWt, 512, 256);

  // 2) fused expert MLP (192 KB dynamic LDS per workgroup; 320 KB available)
  const size_t ldsMoe = (size_t)(64 * 512 + 64 * 1024) * sizeof(unsigned short);
  hipFuncSetAttribute((const void*)k_moe_fused,
                      hipFuncAttributeMaxDynamicSharedMemorySize, (int)ldsMoe);
  k_moe_fused<<<dim3(B_TOT / 64, NEXP), 256, ldsMoe, stream>>>(
      x, Wt0, Wt1, Wt2, Wt3, b0, b1, b2, b3, out_e);

  // 3) gate hidden (WMMA), logits+softmax, weighted combine
  const size_t ldsGate = (size_t)(64 * OBS) * sizeof(unsigned short);
  k_gate_hidden<<<dim3(B_TOT / 64), 256, ldsGate, stream>>>(x, gWt, gb0, g);
  k_gate_softmax<<<dim3(B_TOT / 256), 256, 0, stream>>>(g, gW1, gb1, wsoft);
  k_combine<<<dim3((B_TOT * ACTD) / 256), 256, 0, stream>>>(out_e, wsoft, (float*)d_out);
}
